// PANet_21912923144438
// MI455X (gfx1250) — matrix operand order repaired and checked by reading, not compiled, not run on hardware
//
#include <hip/hip_runtime.h>
#include <hip/hip_bf16.h>

// ---- problem dims (from reference setup_inputs) ----
#define B_   32
#define M_   256
#define N_   1024
#define D_   768
#define K_   8
#define INV2W2 50.0f   // 1 / (2 * 0.1^2)

typedef _Float16 half8_t __attribute__((ext_vector_type(8)));
typedef _Float16 v16h    __attribute__((ext_vector_type(16)));
typedef float    v8f     __attribute__((ext_vector_type(8)));

// ---------------------------------------------------------------------------
// WMMA fragment loaders. Both GEMMs have leading dim D_ = 768 halves.
// A 16x32 f16 fragment (ISA 7.12.2): lane<16 -> row M=lane, K = {k0..k0+7, k0+16..k0+23}
//                                    lane>=16 -> row M=lane-16, K = {k0+8.., k0+24..}
// => per-lane: two 16-byte chunks at (k0 + hs*8) and +16 halves.
__device__ __forceinline__ v16h load_a_frag(const _Float16* __restrict__ A,
                                            int row, int k0, int hs) {
  const _Float16* p = A + (size_t)row * D_ + k0 + hs * 8;
  half8_t c0 = *(const half8_t*)(p);
  half8_t c1 = *(const half8_t*)(p + 16);
  v16h a;
#pragma unroll
  for (int i = 0; i < 8; ++i) { a[i] = c0[i]; a[8 + i] = c1[i]; }
  return a;
}

// B 32x16 f16 fragment: lane<16 -> col N=lane, K = {k0..k0+7, k0+16..k0+23} ; lane>=16 -> col N=lane-16,
// K = {k0+8.., k0+24..}. Source matrix is stored row-major [col, K] (K contiguous), so each
// lane reads two 16-byte chunks.
__device__ __forceinline__ v16h load_b_frag(const _Float16* __restrict__ Bm,
                                            int col, int k0, int hs) {
  const _Float16* p = Bm + (size_t)col * D_ + k0 + hs * 8;
  half8_t c0 = *(const half8_t*)(p);
  half8_t c1 = *(const half8_t*)(p + 16);
  v16h b;
#pragma unroll
  for (int i = 0; i < 8; ++i) { b[i] = c0[i]; b[8 + i] = c1[i]; }
  return b;
}

// ---------------------------------------------------------------------------
// Row L2-normalize + f16 convert. One 256-thread block per row of 768 floats.
// Optionally also writes the raw (unnormalized) f16 copy (for attention GEMM A).
__global__ __launch_bounds__(256) void panet_norm_kernel(
    const float* __restrict__ x, _Float16* __restrict__ xn,
    _Float16* __restrict__ xraw) {
  __shared__ float red[256];
  const int row = blockIdx.x;
  const int tid = threadIdx.x;
  const float* xp = x + (size_t)row * D_;
  const float v0 = xp[tid], v1 = xp[tid + 256], v2 = xp[tid + 512];
  red[tid] = v0 * v0 + v1 * v1 + v2 * v2;
  __syncthreads();
  for (int s = 128; s > 0; s >>= 1) {
    if (tid < s) red[tid] += red[tid + s];
    __syncthreads();
  }
  const float scale = 1.0f / fmaxf(sqrtf(red[0]), 1e-12f);
  _Float16* op = xn + (size_t)row * D_;
  op[tid]       = (_Float16)(v0 * scale);
  op[tid + 256] = (_Float16)(v1 * scale);
  op[tid + 512] = (_Float16)(v2 * scale);
  if (xraw) {
    _Float16* rp = xraw + (size_t)row * D_;
    rp[tid]       = (_Float16)v0;
    rp[tid + 256] = (_Float16)v1;
    rp[tid + 512] = (_Float16)v2;
  }
}

__global__ void panet_conv_half_kernel(const float* __restrict__ x,
                                       _Float16* __restrict__ y, int n) {
  const int i = blockIdx.x * 256 + threadIdx.x;
  if (i < n) y[i] = (_Float16)x[i];
}

__global__ void panet_zero_kernel(float* __restrict__ p, int n) {
  const int i = blockIdx.x * 256 + threadIdx.x;
  if (i < n) p[i] = 0.0f;
}

// ---------------------------------------------------------------------------
// GEMM1 (cosine similarity) + fused mask + RBF kernel pooling into Kfeat[B,M,K].
// Block: 256 thr = 8 waves, each wave -> 32x32 tile; block tile = 64(m) x 128(n).
__global__ __launch_bounds__(256) void panet_gemm_rbf_kernel(
    const _Float16* __restrict__ p_h, const _Float16* __restrict__ e_h,
    const int* __restrict__ pmask, const int* __restrict__ emask,
    const float* __restrict__ mu, float* __restrict__ Kfeat) {
  const int b    = blockIdx.z;
  const int m0   = blockIdx.y * 64;
  const int n0   = blockIdx.x * 128;
  const int lane = threadIdx.x & 31;
  const int wave = threadIdx.x >> 5;
  const int wm   = wave & 1;
  const int wn   = wave >> 1;
  const int hs   = lane >> 4;
  const int l15  = lane & 15;
  const int mw   = m0 + wm * 32;
  const int nw   = n0 + wn * 32;

  const _Float16* A  = p_h + (size_t)b * M_ * D_;
  const _Float16* Bm = e_h + (size_t)b * N_ * D_;

  float mu_r[K_];
#pragma unroll
  for (int k = 0; k < K_; ++k) mu_r[k] = mu[k];

  v8f acc[4];
#pragma unroll
  for (int t = 0; t < 4; ++t)
#pragma unroll
    for (int e = 0; e < 8; ++e) acc[t][e] = 0.0f;

  for (int k0 = 0; k0 < D_; k0 += 32) {
    v16h a0 = load_a_frag(A, mw + l15,      k0, hs);
    v16h a1 = load_a_frag(A, mw + 16 + l15, k0, hs);
    v16h b0 = load_b_frag(Bm, nw + l15,      k0, hs);
    v16h b1 = load_b_frag(Bm, nw + 16 + l15, k0, hs);
    acc[0] = __builtin_amdgcn_wmma_f32_16x16x32_f16(false, a0, false, b0, (short)0, acc[0], false, false);
    acc[1] = __builtin_amdgcn_wmma_f32_16x16x32_f16(false, a0, false, b1, (short)0, acc[1], false, false);
    acc[2] = __builtin_amdgcn_wmma_f32_16x16x32_f16(false, a1, false, b0, (short)0, acc[2], false, false);
    acc[3] = __builtin_amdgcn_wmma_f32_16x16x32_f16(false, a1, false, b1, (short)0, acc[3], false, false);
  }

  // epilogue: mask, RBF, reduce over n within the wave, atomic into Kfeat
  const float em0 = (float)emask[b * N_ + nw + l15];
  const float em1 = (float)emask[b * N_ + nw + 16 + l15];

#pragma unroll
  for (int im = 0; im < 2; ++im) {
#pragma unroll
    for (int r = 0; r < 8; ++r) {
      const int mrow   = mw + im * 16 + hs * 8 + r;  // C layout: M = r + 8*hs
      const float pmv  = (float)pmask[b * M_ + mrow];
      const float v0   = acc[im * 2 + 0][r] * pmv * em0;
      const float v1   = acc[im * 2 + 1][r] * pmv * em1;
#pragma unroll
      for (int k = 0; k < K_; ++k) {
        const float d0 = v0 - mu_r[k];
        const float d1 = v1 - mu_r[k];
        float s = __expf(-INV2W2 * d0 * d0) + __expf(-INV2W2 * d1 * d1);
        s += __shfl_xor(s, 8);
        s += __shfl_xor(s, 4);
        s += __shfl_xor(s, 2);
        s += __shfl_xor(s, 1);
        if (l15 == 0) atomicAdd(&Kfeat[(size_t)(b * M_ + mrow) * K_ + k], s);
      }
    }
  }
}

// ---------------------------------------------------------------------------
// GEMM2: u = P @ W^T, fused tanh + dot with ctx -> scores[B*M] via atomics.
// Rows = B*M = 8192 (64/block), cols = D_ = 768 (128/block).
__global__ __launch_bounds__(256) void panet_gemm_attn_kernel(
    const _Float16* __restrict__ pr_h, const _Float16* __restrict__ w_h,
    const float* __restrict__ b_attn, const float* __restrict__ ctx,
    float* __restrict__ scores) {
  const int g0   = blockIdx.y * 64;   // global row base (b*M + m)
  const int n0   = blockIdx.x * 128;  // j base
  const int lane = threadIdx.x & 31;
  const int wave = threadIdx.x >> 5;
  const int wm   = wave & 1;
  const int wn   = wave >> 1;
  const int hs   = lane >> 4;
  const int l15  = lane & 15;
  const int rw   = g0 + wm * 32;
  const int jw   = n0 + wn * 32;

  v8f acc[4];
#pragma unroll
  for (int t = 0; t < 4; ++t)
#pragma unroll
    for (int e = 0; e < 8; ++e) acc[t][e] = 0.0f;

  for (int k0 = 0; k0 < D_; k0 += 32) {
    v16h a0 = load_a_frag(pr_h, rw + l15,      k0, hs);
    v16h a1 = load_a_frag(pr_h, rw + 16 + l15, k0, hs);
    v16h b0 = load_b_frag(w_h, jw + l15,      k0, hs);
    v16h b1 = load_b_frag(w_h, jw + 16 + l15, k0, hs);
    acc[0] = __builtin_amdgcn_wmma_f32_16x16x32_f16(false, a0, false, b0, (short)0, acc[0], false, false);
    acc[1] = __builtin_amdgcn_wmma_f32_16x16x32_f16(false, a0, false, b1, (short)0, acc[1], false, false);
    acc[2] = __builtin_amdgcn_wmma_f32_16x16x32_f16(false, a1, false, b0, (short)0, acc[2], false, false);
    acc[3] = __builtin_amdgcn_wmma_f32_16x16x32_f16(false, a1, false, b1, (short)0, acc[3], false, false);
  }

  const int j0 = jw + l15, j1 = jw + 16 + l15;
  const float bj0 = b_attn[j0], bj1 = b_attn[j1];
  const float cj0 = ctx[j0],    cj1 = ctx[j1];

#pragma unroll
  for (int im = 0; im < 2; ++im) {
#pragma unroll
    for (int r = 0; r < 8; ++r) {
      const int grow = rw + im * 16 + hs * 8 + r;
      float s = tanhf(acc[im * 2 + 0][r] + bj0) * cj0 +
                tanhf(acc[im * 2 + 1][r] + bj1) * cj1;
      s += __shfl_xor(s, 8);
      s += __shfl_xor(s, 4);
      s += __shfl_xor(s, 2);
      s += __shfl_xor(s, 1);
      if (l15 == 0) atomicAdd(&scores[grow], s);
    }
  }
}

// ---------------------------------------------------------------------------
// Softmax over m + max/min/avg pooling -> out[B, 3K]. One block per batch.
__global__ __launch_bounds__(256) void panet_pool_kernel(
    const float* __restrict__ Kfeat, const float* __restrict__ scores,
    const int* __restrict__ pmask, float* __restrict__ out) {
  __shared__ float red[256];
  const int b = blockIdx.x;
  const int m = threadIdx.x;
  const int valid = pmask[b * M_ + m];
  const float NEG_INF = __int_as_float(0xff800000);
  const float s = valid ? scores[b * M_ + m] : NEG_INF;

  // softmax max
  red[m] = s; __syncthreads();
  for (int t = 128; t > 0; t >>= 1) {
    if (m < t) red[m] = fmaxf(red[m], red[m + t]);
    __syncthreads();
  }
  const float smax = red[0]; __syncthreads();
  const float e = valid ? __expf(s - smax) : 0.0f;
  red[m] = e; __syncthreads();
  for (int t = 128; t > 0; t >>= 1) {
    if (m < t) red[m] += red[m + t];
    __syncthreads();
  }
  const float attn = e / red[0]; __syncthreads();

  float kf[K_];
#pragma unroll
  for (int k = 0; k < K_; ++k) kf[k] = Kfeat[(size_t)(b * M_ + m) * K_ + k];

  for (int k = 0; k < K_; ++k) {
    red[m] = kf[k]; __syncthreads();
    for (int t = 128; t > 0; t >>= 1) {
      if (m < t) red[m] = fmaxf(red[m], red[m + t]);
      __syncthreads();
    }
    if (m == 0) out[b * 24 + k] = red[0];
    __syncthreads();

    red[m] = kf[k]; __syncthreads();
    for (int t = 128; t > 0; t >>= 1) {
      if (m < t) red[m] = fminf(red[m], red[m + t]);
      __syncthreads();
    }
    if (m == 0) out[b * 24 + 8 + k] = red[0];
    __syncthreads();

    red[m] = attn * kf[k]; __syncthreads();
    for (int t = 128; t > 0; t >>= 1) {
      if (m < t) red[m] += red[m + t];
      __syncthreads();
    }
    if (m == 0) out[b * 24 + 16 + k] = red[0];
    __syncthreads();
  }
}

// ---------------------------------------------------------------------------
extern "C" void kernel_launch(void* const* d_in, const int* in_sizes, int n_in,
                              void* d_out, int out_size, void* d_ws, size_t ws_size,
                              hipStream_t stream) {
  const float* prompt = (const float*)d_in[0];  // [B,M,D]
  const float* essay  = (const float*)d_in[1];  // [B,N,D]
  const int*   pmask  = (const int*)d_in[2];    // [B,M]
  const int*   emask  = (const int*)d_in[3];    // [B,N]
  const float* mu     = (const float*)d_in[4];  // [K]
  const float* W      = (const float*)d_in[5];  // [D,D]
  const float* b_attn = (const float*)d_in[6];  // [D]
  const float* ctx    = (const float*)d_in[7];  // [D]
  float* out = (float*)d_out;                   // [B, 3K]

  // workspace layout (all offsets 256B-aligned)
  char* ws = (char*)d_ws;
  _Float16* p_h    = (_Float16*)(ws);                 // 12,582,912 B
  _Float16* pr_h   = (_Float16*)(ws + 12582912);      // 12,582,912 B
  _Float16* e_h    = (_Float16*)(ws + 25165824);      // 50,331,648 B
  _Float16* w_h    = (_Float16*)(ws + 75497472);      //  1,179,648 B
  float*    Kfeat  = (float*)   (ws + 76677120);      //    262,144 B
  float*    scores = (float*)   (ws + 76939264);      //     32,768 B  (contiguous after Kfeat)

  // 1) zero accumulators (Kfeat + scores contiguous: 73728 floats)
  panet_zero_kernel<<<288, 256, 0, stream>>>(Kfeat, B_ * M_ * K_ + B_ * M_);

  // 2) normalize + f16 conversions
  panet_norm_kernel<<<B_ * M_, 256, 0, stream>>>(prompt, p_h, pr_h);
  panet_norm_kernel<<<B_ * N_, 256, 0, stream>>>(essay, e_h, (_Float16*)nullptr);
  panet_conv_half_kernel<<<(D_ * D_ + 255) / 256, 256, 0, stream>>>(W, w_h, D_ * D_);

  // 3) cosine-sim GEMM + fused RBF pooling
  panet_gemm_rbf_kernel<<<dim3(N_ / 128, M_ / 64, B_), 256, 0, stream>>>(
      p_h, e_h, pmask, emask, mu, Kfeat);

  // 4) attention GEMM + fused tanh/ctx reduction
  panet_gemm_attn_kernel<<<dim3(D_ / 128, (B_ * M_) / 64), 256, 0, stream>>>(
      pr_h, w_h, b_attn, ctx, scores);

  // 5) softmax + pooling
  panet_pool_kernel<<<B_, 256, 0, stream>>>(Kfeat, scores, pmask, out);
}
